// MultiheadAttention_18940805775713
// MI455X (gfx1250) — matrix operand order repaired and checked by reading, not compiled, not run on hardware
//
#include <hip/hip_runtime.h>
#include <hip/hip_bf16.h>
#include <stdint.h>

typedef __bf16 bf16_t;
typedef __attribute__((ext_vector_type(16))) __bf16 v16bf;
typedef __attribute__((ext_vector_type(8)))  __bf16 v8bf;
typedef __attribute__((ext_vector_type(4)))  __bf16 v4bf;
typedef __attribute__((ext_vector_type(2)))  __bf16 v2bf;
typedef __attribute__((ext_vector_type(8)))  float  v8f;

union U16bf { v16bf v; v8bf h[2]; };

// Async global -> LDS copy of 16 bytes (CDNA5 ASYNCcnt path, bypasses VGPRs).
__device__ __forceinline__ void async_copy_b128(void* lds_ptr, const void* gptr) {
    uint32_t lds = (uint32_t)(uintptr_t)lds_ptr;      // low 32 bits of generic = LDS offset
    uint64_t ga  = (uint64_t)(uintptr_t)gptr;
    asm volatile("global_load_async_to_lds_b128 %0, %1, off" :: "v"(lds), "v"(ga) : "memory");
}
__device__ __forceinline__ void wait_async0() {
    asm volatile("s_wait_asynccnt 0x0" ::: "memory");
}

// ---------------------------------------------------------------------------
// Prep kernels
// ---------------------------------------------------------------------------
__global__ void cvt_bf16_kernel(const float* __restrict__ in, bf16_t* __restrict__ out, int n) {
    int i = (blockIdx.x * blockDim.x + threadIdx.x) * 4;
    if (i < n) {
        float4 f = *(const float4*)(in + i);
        v4bf o4 = {(bf16_t)f.x, (bf16_t)f.y, (bf16_t)f.z, (bf16_t)f.w};
        *(v4bf*)(out + i) = o4;
    }
}

__global__ void cvt_T_bf16_kernel(const float* __restrict__ W, bf16_t* __restrict__ Wt, int K, int N) {
    int i = blockIdx.x * blockDim.x + threadIdx.x;
    if (i < K * N) {
        int kk = i / N, nn = i - kk * N;
        Wt[(size_t)nn * K + kk] = (bf16_t)W[i];
    }
}

// ---------------------------------------------------------------------------
// NT GEMM: C[M,N] = (A[M,K] * Bt[N,K]^T + bias) * cscale. bf16 in, f32 accum.
// 128x128 block tile, TK=32, 8 waves (2 M x 4 N), wave computes 64x32
// (4x2 WMMA -> 8 WMMAs per 6 fragment loads). Double-buffered LDS fed by
// global_load_async_to_lds_b128; one barrier per K-step.
// ---------------------------------------------------------------------------
template <bool OUT_BF16>
__global__ __launch_bounds__(256) void gemm_nt_kernel(
    const bf16_t* __restrict__ A, const bf16_t* __restrict__ Bt,
    const float* __restrict__ bias, void* __restrict__ Cout,
    int M, int N, int K, float cscale)
{
    __shared__ alignas(16) bf16_t Alds[2][128][40];    // rows 80B, 16B aligned
    __shared__ alignas(16) bf16_t Blds[2][128][40];

    const int tid  = threadIdx.x;
    const int wid  = tid >> 5;
    const int lane = tid & 31;
    const int half = lane & 15;
    const int hi   = lane >> 4;
    const int wm   = wid & 1;        // 0..1 -> 64-row half
    const int wn   = wid >> 1;       // 0..3 -> 32-col strip
    const int m0   = blockIdx.x * 128;
    const int n0   = blockIdx.y * 128;

    auto prefetch = [&](int buf, int k0) {
#pragma unroll
        for (int i = 0; i < 2; ++i) {                  // A: 128x32 = 512 chunks
            int cc = tid * 2 + i;
            int r = cc >> 2, c = (cc & 3) * 8;
            async_copy_b128(&Alds[buf][r][c], A + (size_t)(m0 + r) * K + k0 + c);
        }
#pragma unroll
        for (int i = 0; i < 2; ++i) {                  // B: 128x32
            int cc = tid * 2 + i;
            int r = cc >> 2, c = (cc & 3) * 8;
            async_copy_b128(&Blds[buf][r][c], Bt + (size_t)(n0 + r) * K + k0 + c);
        }
    };

    v8f acc[4][2] = {};
    prefetch(0, 0);
    const int KT = K >> 5;
    for (int kt = 0; kt < KT; ++kt) {
        wait_async0();
        __syncthreads();
        if (kt + 1 < KT) prefetch((kt + 1) & 1, (kt + 1) << 5);
        const int buf = kt & 1;

        // Hoist ALL fragment loads so ds_load_b128s batch ahead of the WMMAs
        U16bf afr[4], bfr[2];
#pragma unroll
        for (int mt = 0; mt < 4; ++mt) {
            int row = wm * 64 + mt * 16 + half;
            afr[mt].h[0] = *(const v8bf*)&Alds[buf][row][hi * 8];
            afr[mt].h[1] = *(const v8bf*)&Alds[buf][row][16 + hi * 8];
        }
#pragma unroll
        for (int nt = 0; nt < 2; ++nt) {
            int row = wn * 32 + nt * 16 + half;
            bfr[nt].h[0] = *(const v8bf*)&Blds[buf][row][hi * 8];
            bfr[nt].h[1] = *(const v8bf*)&Blds[buf][row][16 + hi * 8];
        }
#pragma unroll
        for (int mt = 0; mt < 4; ++mt)
#pragma unroll
            for (int nt = 0; nt < 2; ++nt)
                acc[mt][nt] = __builtin_amdgcn_wmma_f32_16x16x32_bf16(
                    false, afr[mt].v, false, bfr[nt].v, (short)0, acc[mt][nt], false, false);
    }

#pragma unroll
    for (int mt = 0; mt < 4; ++mt)
#pragma unroll
        for (int nt = 0; nt < 2; ++nt) {
            int n = n0 + wn * 32 + nt * 16 + half;
            float bv = bias[n];
            int mbase = m0 + wm * 64 + mt * 16 + hi * 8;
#pragma unroll
            for (int r = 0; r < 8; ++r) {
                float val = (acc[mt][nt][r] + bv) * cscale;
                size_t off = (size_t)(mbase + r) * N + n;
                if (OUT_BF16) ((bf16_t*)Cout)[off] = (bf16_t)val;
                else          ((float*)Cout)[off]  = val;
            }
        }
}

// ---------------------------------------------------------------------------
// Flash attention. Block = (b*H+h, 128-query tile), 256 threads = 8 waves,
// wave owns 16 q rows. Key tiles of 64 amortized over 128 q-rows. Q is
// pre-scaled by 1/sqrt(Dh) in the Q-projection GEMM. K tiles arrive via
// async-to-LDS (double buffered); V is register-staged one tile ahead and
// transposed into LDS with packed b32 stores. 16 WMMA per wave per key tile.
// ---------------------------------------------------------------------------
__global__ __launch_bounds__(256) void attn_kernel(
    const bf16_t* __restrict__ q, const bf16_t* __restrict__ kk,
    const bf16_t* __restrict__ v, bf16_t* __restrict__ att,
    int Bn, int S, int H)
{
    const int Dh = 64;
    const int D  = H * Dh;

    __shared__ alignas(16) bf16_t Klds[2][64][72];   // 64 keys x 64 dh, rows 144B
    __shared__ alignas(16) bf16_t Vlds[2][64][72];   // dh(64) x keys(64), transposed
    __shared__ alignas(16) bf16_t Plds[8][16][72];   // per-wave P staging

    const int tid  = threadIdx.x;
    const int wid  = tid >> 5;
    const int lane = tid & 31;
    const int half = lane & 15;
    const int hi   = lane >> 4;
    const int bh   = blockIdx.y;
    const int b    = bh / H, h = bh % H;
    const int q0   = blockIdx.x * 128 + wid * 16;

    // Q fragments (A layout) for the whole key loop; Q already scaled by 1/8.
    U16bf qf[2];
    {
        const bf16_t* qrow = q + ((size_t)b * S + q0 + half) * D + h * Dh;
#pragma unroll
        for (int kc = 0; kc < 2; ++kc) {
            qf[kc].h[0] = *(const v8bf*)(qrow + kc * 32 + hi * 8);
            qf[kc].h[1] = *(const v8bf*)(qrow + kc * 32 + 16 + hi * 8);
        }
    }

    auto prefetchK = [&](int buf, int kt) {
#pragma unroll
        for (int i = 0; i < 2; ++i) {
            int cc = tid * 2 + i;               // 512 chunks: 64 rows x 8
            int r = cc >> 3, p = cc & 7;
            async_copy_b128(&Klds[buf][r][p * 8],
                            kk + ((size_t)b * S + kt + r) * D + h * Dh + p * 8);
        }
    };
    v8bf vs[2];                                  // staged V: one key-pair chunk
    const int vr2 = tid >> 3, vp = tid & 7;      // 256 pair-chunks: 32 key-pairs x 8
    auto loadVreg = [&](int kt) {
        const bf16_t* base = v + ((size_t)b * S + kt + 2 * vr2) * D + h * Dh + vp * 8;
        vs[0] = *(const v8bf*)(base);
        vs[1] = *(const v8bf*)(base + D);
    };
    auto storeV = [&](int buf) {
#pragma unroll
        for (int e = 0; e < 8; ++e) {
            v2bf t; t[0] = vs[0][e]; t[1] = vs[1][e];
            *(v2bf*)&Vlds[buf][vp * 8 + e][2 * vr2] = t;   // packed ds_store_b32
        }
    };

    v8f o[4] = {};
    float mrow[8], lrow[8];
#pragma unroll
    for (int r = 0; r < 8; ++r) { mrow[r] = -__builtin_inff(); lrow[r] = 0.0f; }

    prefetchK(0, 0);
    loadVreg(0);
    const int NT = S / 64;
    for (int t = 0; t < NT; ++t) {
        const int buf = t & 1;
        wait_async0();
        storeV(buf);
        __syncthreads();
        if (t + 1 < NT) { prefetchK(buf ^ 1, (t + 1) * 64); loadVreg((t + 1) * 64); }

        // ---- Scores: hoist all 8 K-fragments, then 8 WMMAs -----------------
        U16bf kf[4][2];
#pragma unroll
        for (int nt = 0; nt < 4; ++nt)
#pragma unroll
            for (int kc = 0; kc < 2; ++kc) {
                kf[nt][kc].h[0] = *(const v8bf*)&Klds[buf][nt * 16 + half][kc * 32 + hi * 8];
                kf[nt][kc].h[1] = *(const v8bf*)&Klds[buf][nt * 16 + half][kc * 32 + 16 + hi * 8];
            }
        v8f sc[4];
#pragma unroll
        for (int nt = 0; nt < 4; ++nt) {
            v8f a = {};
            a = __builtin_amdgcn_wmma_f32_16x16x32_bf16(
                false, qf[0].v, false, kf[nt][0].v, (short)0, a, false, false);
            a = __builtin_amdgcn_wmma_f32_16x16x32_bf16(
                false, qf[1].v, false, kf[nt][1].v, (short)0, a, false, false);
            sc[nt] = a;
        }

        // ---- Online softmax (C layout: row = hi*8+r over 16 lanes) ---------
        float mnew[8], alpha[8];
#pragma unroll
        for (int r = 0; r < 8; ++r) {
            float mloc = fmaxf(fmaxf(sc[0][r], sc[1][r]), fmaxf(sc[2][r], sc[3][r]));
#pragma unroll
            for (int off2 = 1; off2 < 16; off2 <<= 1)
                mloc = fmaxf(mloc, __shfl_xor(mloc, off2, 32));
            mnew[r]  = fmaxf(mrow[r], mloc);
            alpha[r] = __expf(mrow[r] - mnew[r]);
            mrow[r]  = mnew[r];
        }
#pragma unroll
        for (int r = 0; r < 8; ++r) {
            float s = 0.0f;
#pragma unroll
            for (int nt = 0; nt < 4; ++nt) {
                sc[nt][r] = __expf(sc[nt][r] - mnew[r]);
                s += sc[nt][r];
            }
#pragma unroll
            for (int off2 = 1; off2 < 16; off2 <<= 1) s += __shfl_xor(s, off2, 32);
            lrow[r] = lrow[r] * alpha[r] + s;
        }
#pragma unroll
        for (int d4 = 0; d4 < 4; ++d4)
#pragma unroll
            for (int r = 0; r < 8; ++r) o[d4][r] *= alpha[r];

        // ---- P: C layout -> A layout via per-wave LDS staging --------------
#pragma unroll
        for (int nt = 0; nt < 4; ++nt)
#pragma unroll
            for (int r = 0; r < 8; ++r)
                Plds[wid][hi * 8 + r][nt * 16 + half] = (bf16_t)sc[nt][r];

        U16bf pf[2];
#pragma unroll
        for (int kc = 0; kc < 2; ++kc) {
            pf[kc].h[0] = *(const v8bf*)&Plds[wid][half][kc * 32 + hi * 8];
            pf[kc].h[1] = *(const v8bf*)&Plds[wid][half][kc * 32 + 16 + hi * 8];
        }

        // ---- O += P * V : hoist all 8 V-fragments, then 8 WMMAs ------------
        U16bf vf[4][2];
#pragma unroll
        for (int d4 = 0; d4 < 4; ++d4)
#pragma unroll
            for (int kc = 0; kc < 2; ++kc) {
                vf[d4][kc].h[0] = *(const v8bf*)&Vlds[buf][d4 * 16 + half][kc * 32 + hi * 8];
                vf[d4][kc].h[1] = *(const v8bf*)&Vlds[buf][d4 * 16 + half][kc * 32 + 16 + hi * 8];
            }
#pragma unroll
        for (int d4 = 0; d4 < 4; ++d4) {
            o[d4] = __builtin_amdgcn_wmma_f32_16x16x32_bf16(
                false, pf[0].v, false, vf[d4][0].v, (short)0, o[d4], false, false);
            o[d4] = __builtin_amdgcn_wmma_f32_16x16x32_bf16(
                false, pf[1].v, false, vf[d4][1].v, (short)0, o[d4], false, false);
        }
    }

    // Normalize + store bf16
#pragma unroll
    for (int d4 = 0; d4 < 4; ++d4)
#pragma unroll
        for (int r = 0; r < 8; ++r) {
            float val = o[d4][r] / lrow[r];
            size_t row = (size_t)b * S + q0 + hi * 8 + r;
            att[row * D + h * Dh + d4 * 16 + half] = (bf16_t)val;
        }
}

// ---------------------------------------------------------------------------
extern "C" void kernel_launch(void* const* d_in, const int* in_sizes, int n_in,
                              void* d_out, int out_size, void* d_ws, size_t ws_size,
                              hipStream_t stream)
{
    (void)in_sizes; (void)n_in; (void)out_size; (void)ws_size;
    const int Bn = 4, S = 2048, D = 1024, H = 16;
    const int M = Bn * S;

    const float* x  = (const float*)d_in[0];
    const float* Wq = (const float*)d_in[1];
    const float* bq = (const float*)d_in[2];
    const float* Wk = (const float*)d_in[3];
    const float* bk = (const float*)d_in[4];
    const float* Wv = (const float*)d_in[5];
    const float* bv = (const float*)d_in[6];
    const float* Wo = (const float*)d_in[7];
    const float* bo = (const float*)d_in[8];

    char* ws = (char*)d_ws;
    size_t off = 0;
    auto alloc = [&](size_t bytes) -> char* {
        char* p = ws + off;
        off += (bytes + 255) & ~(size_t)255;
        return p;
    };
    bf16_t* xb   = (bf16_t*)alloc((size_t)M * D * sizeof(bf16_t));
    bf16_t* wqT  = (bf16_t*)alloc((size_t)D * D * sizeof(bf16_t));
    bf16_t* wkT  = (bf16_t*)alloc((size_t)D * D * sizeof(bf16_t));
    bf16_t* wvT  = (bf16_t*)alloc((size_t)D * D * sizeof(bf16_t));
    bf16_t* woT  = (bf16_t*)alloc((size_t)D * D * sizeof(bf16_t));
    bf16_t* qb   = (bf16_t*)alloc((size_t)M * D * sizeof(bf16_t));
    bf16_t* kb   = (bf16_t*)alloc((size_t)M * D * sizeof(bf16_t));
    bf16_t* vb   = (bf16_t*)alloc((size_t)M * D * sizeof(bf16_t));
    bf16_t* attb = (bf16_t*)alloc((size_t)M * D * sizeof(bf16_t));

    {
        int n = M * D;
        cvt_bf16_kernel<<<(n / 4 + 255) / 256, 256, 0, stream>>>(x, xb, n);
        int nw = D * D;
        cvt_T_bf16_kernel<<<(nw + 255) / 256, 256, 0, stream>>>(Wq, wqT, D, D);
        cvt_T_bf16_kernel<<<(nw + 255) / 256, 256, 0, stream>>>(Wk, wkT, D, D);
        cvt_T_bf16_kernel<<<(nw + 255) / 256, 256, 0, stream>>>(Wv, wvT, D, D);
        cvt_T_bf16_kernel<<<(nw + 255) / 256, 256, 0, stream>>>(Wo, woT, D, D);
    }

    dim3 gg(M / 128, D / 128);
    // Fold 1/sqrt(Dh)=0.125 into the Q projection (removes inner-loop scaling)
    gemm_nt_kernel<true><<<gg, 256, 0, stream>>>(xb, wqT, bq, (void*)qb, M, D, D, 0.125f);
    gemm_nt_kernel<true><<<gg, 256, 0, stream>>>(xb, wkT, bk, (void*)kb, M, D, D, 1.0f);
    gemm_nt_kernel<true><<<gg, 256, 0, stream>>>(xb, wvT, bv, (void*)vb, M, D, D, 1.0f);

    dim3 ga(S / 128, Bn * H);
    attn_kernel<<<ga, 256, 0, stream>>>(qb, kb, vb, attb, Bn, S, H);

    gemm_nt_kernel<false><<<gg, 256, 0, stream>>>(attb, woT, bo, d_out, M, D, D, 1.0f);
}
